// GAT_20993800142885
// MI455X (gfx1250) — compile-verified, matrix operand order checked by reading, not run on hardware
//
#include <hip/hip_runtime.h>
#include <cstdint>
#include <cstddef>

typedef __attribute__((ext_vector_type(16))) _Float16 v16h;
typedef __attribute__((ext_vector_type(8)))  float    v8f;

#define NEG_SLOPE 0.2f

// ---------------------------------------------------------------------------
// Generic 32-bit fill (init atomic accumulators each launch)
// ---------------------------------------------------------------------------
__global__ void k_fill_u32(uint32_t* __restrict__ p, uint32_t v, long long n) {
  long long i = (long long)blockIdx.x * blockDim.x + threadIdx.x;
  if (i < n) p[i] = v;
}

// ---------------------------------------------------------------------------
// Y[M,N] = X[M,K] * W[N,K]^T   (fp32 in/out, f16 WMMA core, f32 accumulate)
// One wave computes a 16(M) x (16*NT)(N) strip; K consumed in chunks of 32.
// NT chosen so each activation row is loaded exactly once per GEMM.
// A fragment (16x32 f16, ISA 7.12.2): lane&15 = row M, lane>>4 + VGPR idx
// select K. B fragment mirrors A with lane&15 = column N (W is [N,K] row-major).
// C/D: VGPR r -> M = (lane>>4)*8 + r, N = lane&15.
// ---------------------------------------------------------------------------
template <int NT>
__global__ void k_gemm_wmma(const float* __restrict__ X, const float* __restrict__ W,
                            float* __restrict__ Y, int M, int N, int K) {
  const int SW = 16 * NT;                       // strip width
  const int strips = (N + SW - 1) / SW;
  const int wid  = (int)(((long long)blockIdx.x * blockDim.x + threadIdx.x) >> 5);
  const int lane = threadIdx.x & 31;
  const int mtiles = M >> 4;
  if (wid >= mtiles * strips) return;           // wave-uniform: EXEC all-1s for WMMA
  const int m0  = (wid / strips) << 4;
  const int s0  = (wid % strips) * SW;
  const int row = lane & 15;
  const int hi  = lane >> 4;

  v8f acc[NT] = {};
  const float* xr = X + (long long)(m0 + row) * K;

  for (int kc = 0; kc < K; kc += 32) {
    v16h a;
#pragma unroll
    for (int v = 0; v < 8; ++v) {
      int kk = kc + ((v >> 2) << 4) + (hi << 3) + ((v & 3) << 1);
      a[2 * v]     = (_Float16)xr[kk];
      a[2 * v + 1] = (_Float16)xr[kk + 1];
    }
    if (kc + 32 < K) __builtin_prefetch(xr + kc + 32, 0, 3);  // global_prefetch_b8

#pragma unroll
    for (int nt = 0; nt < NT; ++nt) {
      const int ncol = s0 + nt * 16 + row;
      v16h b;
      if (ncol < N) {
        const float* wr = W + (long long)ncol * K;
#pragma unroll
        for (int v = 0; v < 8; ++v) {
          int kk = kc + ((v >> 2) << 4) + (hi << 3) + ((v & 3) << 1);
          b[2 * v]     = (_Float16)wr[kk];
          b[2 * v + 1] = (_Float16)wr[kk + 1];
        }
      } else {
#pragma unroll
        for (int v = 0; v < 16; ++v) b[v] = (_Float16)0.f;
      }
      acc[nt] = __builtin_amdgcn_wmma_f32_16x16x32_f16(
          /*neg_a=*/false, a, /*neg_b=*/false, b,
          /*c_mod=*/(short)0, acc[nt], /*reuse_a=*/false, /*reuse_b=*/false);
    }
  }

#pragma unroll
  for (int nt = 0; nt < NT; ++nt) {
    const int col = s0 + nt * 16 + row;
    if (col < N) {
#pragma unroll
      for (int r = 0; r < 8; ++r)
        Y[(long long)(m0 + (hi << 3) + r) * N + col] = acc[nt][r];
    }
  }
}

// ---------------------------------------------------------------------------
// Per-node attention scalars: aL[n] = <Xl[n,:], attL>, aR[n] = <Xl[n,:], attR>
// One wave32 per node, shuffle reduction.
// ---------------------------------------------------------------------------
__global__ void k_rowdot(const float* __restrict__ Xl, const float* __restrict__ attL,
                         const float* __restrict__ attR, float* __restrict__ aL,
                         float* __restrict__ aR, int n, int C) {
  int wid  = (int)(((long long)blockIdx.x * blockDim.x + threadIdx.x) >> 5);
  int lane = threadIdx.x & 31;
  if (wid >= n) return;
  const float* r = Xl + (long long)wid * C;
  float sl = 0.f, sr = 0.f;
  for (int c = lane; c < C; c += 32) {
    float v = r[c];
    sl += v * attL[c];
    sr += v * attR[c];
  }
#pragma unroll
  for (int o = 16; o > 0; o >>= 1) {
    sl += __shfl_xor(sl, o, 32);
    sr += __shfl_xor(sr, o, 32);
  }
  if (lane == 0) { aL[wid] = sl; aR[wid] = sr; }
}

// Order-preserving int mapping for float atomicMax (involution)
__device__ __forceinline__ int fmap(float f) {
  int i = __float_as_int(f);
  return i >= 0 ? i : (i ^ 0x7FFFFFFF);
}

// Pass 1: e = leaky_relu(aL[src]+aR[dst]); segment max over dst (int atomicMax)
__global__ void k_edge_max(const int* __restrict__ src, const int* __restrict__ dst,
                           const float* __restrict__ aL, const float* __restrict__ aR,
                           float* __restrict__ ev, int* __restrict__ smax, int E) {
  int e = blockIdx.x * blockDim.x + threadIdx.x;
  if (e >= E) return;
  float v = aL[src[e]] + aR[dst[e]];
  v = v > 0.f ? v : v * NEG_SLOPE;
  ev[e] = v;
  atomicMax(smax + dst[e], fmap(v));
}

// Pass 2: ex = exp(e - max[dst]); denom[dst] += ex
__global__ void k_edge_exp(const int* __restrict__ dst, const int* __restrict__ smax,
                           float* __restrict__ ev, float* __restrict__ denom, int E) {
  int e = blockIdx.x * blockDim.x + threadIdx.x;
  if (e >= E) return;
  int d  = dst[e];
  int mi = smax[d];
  float m = __int_as_float(mi >= 0 ? mi : (mi ^ 0x7FFFFFFF));
  float xv = expf(ev[e] - m);
  ev[e] = xv;
  atomicAdd(denom + d, xv);
}

// Pass 3: one wave per edge; Agg[dst,:] += Xl[src,:] * (ex / max(denom,1e-16))
__global__ void k_edge_agg(const int* __restrict__ src, const int* __restrict__ dst,
                           const float* __restrict__ ev, const float* __restrict__ denom,
                           const float* __restrict__ Xl, float* __restrict__ Agg,
                           int E, int C) {
  long long t = (long long)blockIdx.x * blockDim.x + threadIdx.x;
  int e    = (int)(t >> 5);
  int lane = (int)(t & 31);
  if (e >= E) return;
  int s = src[e], d = dst[e];
  float coeff = ev[e] / fmaxf(denom[d], 1e-16f);
  const float* xs = Xl + (long long)s * C;
  float* od = Agg + (long long)d * C;
  for (int c = lane; c < C; c += 32)
    atomicAdd(od + c, xs[c] * coeff);   // global_atomic_add_f32, resolves in L2
}

// h = elu(agg + bias): one block per node row (blockDim.x == C), no int division
__global__ void k_bias_elu(float* __restrict__ h, const float* __restrict__ b) {
  long long i = (long long)blockIdx.x * blockDim.x + threadIdx.x;
  float v = h[i] + b[threadIdx.x];
  h[i] = v > 0.f ? v : (expf(v) - 1.f);
}

// In-place: out[n,:] = log_softmax(out[n,:] + b).  C in (32,64]; one wave/node.
__global__ void k_bias_logsoftmax(float* __restrict__ out, const float* __restrict__ b,
                                  int n, int C) {
  int wid  = (int)(((long long)blockIdx.x * blockDim.x + threadIdx.x) >> 5);
  int lane = threadIdx.x & 31;
  if (wid >= n) return;
  float* row = out + (long long)wid * C;
  float z0 = row[lane] + b[lane];
  bool has2 = (lane + 32) < C;
  float z1 = has2 ? row[lane + 32] + b[lane + 32] : -3.402823466e38f;
  float m = fmaxf(z0, z1);
#pragma unroll
  for (int o = 16; o > 0; o >>= 1) m = fmaxf(m, __shfl_xor(m, o, 32));
  float s = expf(z0 - m) + (has2 ? expf(z1 - m) : 0.f);
#pragma unroll
  for (int o = 16; o > 0; o >>= 1) s += __shfl_xor(s, o, 32);
  float ls = logf(s);
  row[lane] = z0 - m - ls;
  if (has2) row[lane + 32] = z1 - m - ls;
}

// ---------------------------------------------------------------------------
extern "C" void kernel_launch(void* const* d_in, const int* in_sizes, int n_in,
                              void* d_out, int out_size, void* d_ws, size_t ws_size,
                              hipStream_t stream) {
  (void)n_in; (void)out_size; (void)ws_size;
  const float* x   = (const float*)d_in[0];
  const int*   ei  = (const int*)d_in[1];
  const float* W1  = (const float*)d_in[2];
  const float* al1 = (const float*)d_in[3];
  const float* ar1 = (const float*)d_in[4];
  const float* b1  = (const float*)d_in[5];
  const float* W2  = (const float*)d_in[6];
  const float* al2 = (const float*)d_in[7];
  const float* ar2 = (const float*)d_in[8];
  const float* b2  = (const float*)d_in[9];

  const int IN_C = 256, HID = 128, OUT = 40;
  const int NN = in_sizes[0] / IN_C;   // 100000
  const int E  = in_sizes[1] / 2;      // 1600000
  const int* src = ei;
  const int* dst = ei + E;

  // workspace layout (floats)
  float* xl1   = (float*)d_ws;                 // NN*HID  (reused as xl2)
  float* hbuf  = xl1 + (size_t)NN * HID;       // NN*HID  (agg1 -> h)
  float* aL    = hbuf + (size_t)NN * HID;      // NN
  float* aR    = aL + NN;                      // NN
  int*   smax  = (int*)(aR + NN);              // NN (mapped-int segment max)
  float* denom = (float*)(smax + NN);          // NN
  float* evalb = denom + NN;                   // E
  float* xl2   = xl1;                          // NN*OUT fits in xl1's region
  float* outp  = (float*)d_out;                // NN*OUT (agg2, then final)

  const int BS = 256;
  auto blocks = [](long long t, int bs) { return (unsigned)((t + bs - 1) / bs); };

  // ---------------- Layer 1 ----------------
  {
    long long waves = (long long)(NN / 16);     // NT=8 -> strip = 128 = HID, 1 strip
    k_gemm_wmma<8><<<blocks(waves * 32, BS), BS, 0, stream>>>(x, W1, xl1, NN, HID, IN_C);
  }
  k_rowdot<<<blocks((long long)NN * 32, BS), BS, 0, stream>>>(xl1, al1, ar1, aL, aR, NN, HID);
  k_fill_u32<<<blocks(NN, BS), BS, 0, stream>>>((uint32_t*)smax, 0x80000000u, NN);
  k_fill_u32<<<blocks(NN, BS), BS, 0, stream>>>((uint32_t*)denom, 0u, NN);
  k_fill_u32<<<blocks((long long)NN * HID, BS), BS, 0, stream>>>((uint32_t*)hbuf, 0u, (long long)NN * HID);
  k_edge_max<<<blocks(E, BS), BS, 0, stream>>>(src, dst, aL, aR, evalb, smax, E);
  k_edge_exp<<<blocks(E, BS), BS, 0, stream>>>(dst, smax, evalb, denom, E);
  k_edge_agg<<<blocks((long long)E * 32, BS), BS, 0, stream>>>(src, dst, evalb, denom, xl1, hbuf, E, HID);
  k_bias_elu<<<(unsigned)NN, HID, 0, stream>>>(hbuf, b1);

  // ---------------- Layer 2 ----------------
  {
    long long waves = (long long)(NN / 16);     // NT=3 -> strip = 48 >= 40, 1 strip
    k_gemm_wmma<3><<<blocks(waves * 32, BS), BS, 0, stream>>>(hbuf, W2, xl2, NN, OUT, HID);
  }
  k_rowdot<<<blocks((long long)NN * 32, BS), BS, 0, stream>>>(xl2, al2, ar2, aL, aR, NN, OUT);
  k_fill_u32<<<blocks(NN, BS), BS, 0, stream>>>((uint32_t*)smax, 0x80000000u, NN);
  k_fill_u32<<<blocks(NN, BS), BS, 0, stream>>>((uint32_t*)denom, 0u, NN);
  k_fill_u32<<<blocks((long long)NN * OUT, BS), BS, 0, stream>>>((uint32_t*)outp, 0u, (long long)NN * OUT);
  k_edge_max<<<blocks(E, BS), BS, 0, stream>>>(src, dst, aL, aR, evalb, smax, E);
  k_edge_exp<<<blocks(E, BS), BS, 0, stream>>>(dst, smax, evalb, denom, E);
  k_edge_agg<<<blocks((long long)E * 32, BS), BS, 0, stream>>>(src, dst, evalb, denom, xl2, outp, E, OUT);
  k_bias_logsoftmax<<<blocks((long long)NN * 32, BS), BS, 0, stream>>>(outp, b2, NN, OUT);
}
